// InputAttention_48301202210901
// MI455X (gfx1250) — matrix operand order repaired and checked by reading, not compiled, not run on hardware
//
#include <hip/hip_runtime.h>
#include <hip/hip_bf16.h>

typedef __attribute__((ext_vector_type(16))) __bf16 v16bf;
typedef __attribute__((ext_vector_type(8)))  float  v8f;

#define B_  512
#define T_  512
#define N_  128
#define M_  64
#define U_  64
#define KT  32            // K (t) tile per iteration
#define NKIT (T_ / KT)    // 16 K-steps
#define LXS 34            // padded LDS stride for x chunk  [n][t_local]
#define LWS 34            // padded LDS stride for w1 chunk [u][t_local]

__global__ __launch_bounds__(256)
void InputAttention_48301202210901_kernel(
    const float* __restrict__ x,    // (B,T,N)
    const float* __restrict__ hid,  // (B,M)
    const float* __restrict__ cel,  // (B,M)
    const float* __restrict__ w1,   // (T,U)
    const float* __restrict__ w1b,  // (U)
    const float* __restrict__ w2,   // (2M,U)
    const float* __restrict__ w2b,  // (U)
    const float* __restrict__ vk,   // (U)
    const float* __restrict__ vb,   // (1)
    float* __restrict__ out)        // ctx (B,T) then aw (B,N)
{
    __shared__ __bf16 sx[N_ * LXS];            // x chunk, bf16, [n][t_local]
    __shared__ __bf16 sw[U_ * LWS];            // w1 chunk, bf16, [u][t_local]
    __shared__ float  s_q[U_];                 // q_proj + w1_bias
    __shared__ float  s_v[U_];                 // v_kernel
    __shared__ float  s_z[N_];                 // pre-softmax scores
    __shared__ alignas(16) float s_aw[N_];     // attention weights
    __shared__ float  s_red;

    const int tid  = threadIdx.x;
    const int lane = tid & 31;
    const int wave = tid >> 5;      // 0..7
    const int nb   = wave * 16;     // this wave's n-strip
    const int b    = blockIdx.x;
    const size_t xbase = (size_t)b * T_ * N_;

    // ---------------- Phase A: q_proj = [h;c]@W2 + b2 (+ fold in w1_bias) ----
    if (tid < U_) {
        float acc = w2b[tid];
        const float* h = hid + (size_t)b * M_;
        const float* c = cel + (size_t)b * M_;
        #pragma unroll 8
        for (int j = 0; j < M_; ++j) acc += h[j] * w2[j * U_ + tid];
        #pragma unroll 8
        for (int j = 0; j < M_; ++j) acc += c[j] * w2[(M_ + j) * U_ + tid];
        s_q[tid] = acc + w1b[tid];
        s_v[tid] = vk[tid];
    }

    // ---------------- Phase B: S = x_perm @ W1 via bf16 WMMA ----------------
    v8f cacc[4] = {};   // 4 u-tiles of 16x16 f32 per wave

    // fragment addressing constants (contiguous K: lane half h holds k = 16h..16h+15 for A and B)
    const int m   = lane & 15;               // A: row within n-strip
    const int hb  = (lane >> 4) ? 16 : 0;    // A: lane-half K offset
    const int col = lane & 15;               // B: output column (u)
    const int kh  = (lane >> 4) * 16;        // B: lane-half K offset

    for (int kb = 0; kb < NKIT; ++kb) {
        // stage x chunk (32 t-rows x 128 n) -> LDS transposed as [n][t_local]
        #pragma unroll
        for (int i = 0; i < 16; ++i) {
            int idx = i * 256 + tid;
            int tl  = idx >> 7;        // 0..31
            int n   = idx & 127;
            float v = x[xbase + (size_t)(kb * KT + tl) * N_ + n];
            sx[n * LXS + tl] = (__bf16)v;
        }
        // stage w1 chunk (32 t-rows x 64 u) -> LDS transposed as [u][t_local]
        #pragma unroll
        for (int i = 0; i < 8; ++i) {
            int idx = i * 256 + tid;
            int tl  = idx >> 6;        // 0..31
            int u   = idx & 63;
            float v = w1[(kb * KT + tl) * U_ + u];
            sw[u * LWS + tl] = (__bf16)v;
        }
        // prefetch next x chunk into cache hierarchy (global_prefetch_b8)
        if (kb + 1 < NKIT)
            __builtin_prefetch(&x[xbase + (size_t)((kb + 1) * KT + (tid >> 3)) * N_
                                  + (size_t)(tid & 7) * 16], 0, 1);
        __syncthreads();

        // ---- gather ALL fragments first (one DS clause, one wait), then
        // ---- fire the 4 WMMAs back-to-back on the matrix pipe.
        union { unsigned u[8]; v16bf v; } A;
        union { unsigned u[8]; v16bf v; } Bf[4];

        // A fragment: 16x32 bf16; lanes 0-15: K 0-15,
        //                         lanes 16-31: K 16-31
        #pragma unroll
        for (int r = 0; r < 8; ++r) {
            int k = hb + 2 * r;
            A.u[r] = *(const unsigned*)&sx[(nb + m) * LXS + k];
        }
        // B fragments: 32x16 bf16 per u-tile; lane = column, lane-half = K-half
        #pragma unroll
        for (int ut = 0; ut < 4; ++ut) {
            const __bf16* wr = &sw[(ut * 16 + col) * LWS + kh];
            #pragma unroll
            for (int i = 0; i < 8; ++i) Bf[ut].u[i] = *(const unsigned*)&wr[2 * i];
        }
        #pragma unroll
        for (int ut = 0; ut < 4; ++ut) {
            cacc[ut] = __builtin_amdgcn_wmma_f32_16x16x32_bf16(
                false, A.v, false, Bf[ut].v, (short)0, cacc[ut], false, false);
        }
        __syncthreads();
    }

    // ---------------- Phase C: tanh, dot with v, per-row reduce -------------
    {
        const float vbias = vb[0];
        const int half = lane >> 4;
        #pragma unroll
        for (int r = 0; r < 8; ++r) {
            float p = 0.0f;
            #pragma unroll
            for (int ut = 0; ut < 4; ++ut) {
                int u = ut * 16 + col;
                p += tanhf(cacc[ut][r] + s_q[u]) * s_v[u];
            }
            // reduce the 16 lanes that hold one n-row's 16 u-values
            #pragma unroll
            for (int off = 1; off < 16; off <<= 1)
                p += __shfl_xor(p, off, 32);
            if (col == 0) s_z[nb + half * 8 + r] = p + vbias;
        }
    }
    __syncthreads();

    // ---------------- Phase D: softmax over N=128 ---------------------------
    if (tid == 0) {
        float mx = s_z[0];
        for (int i = 1; i < N_; ++i) mx = fmaxf(mx, s_z[i]);
        s_red = mx;
    }
    __syncthreads();
    if (tid < N_) s_aw[tid] = __expf(s_z[tid] - s_red);
    __syncthreads();
    if (tid == 0) {
        float sm = 0.0f;
        for (int i = 0; i < N_; ++i) sm += s_aw[i];
        s_red = 1.0f / sm;
    }
    __syncthreads();
    if (tid < N_) {
        float a = s_aw[tid] * s_red;
        s_aw[tid] = a;
        out[(size_t)B_ * T_ + (size_t)b * N_ + tid] = a;   // attention_weights
    }
    __syncthreads();

    // ---------------- Phase E: ctx[b,t] = sum_n aw[n] * x[b,t,n] ------------
    // x re-read hits the 192MB L2 (x is 134MB and just streamed through it)
    {
        const float4* awv = (const float4*)s_aw;
        for (int t = tid; t < T_; t += 256) {
            const float4* xr = (const float4*)&x[xbase + (size_t)t * N_];
            float acc = 0.0f;
            #pragma unroll 8
            for (int q = 0; q < N_ / 4; ++q) {
                float4 a4 = awv[q];
                float4 x4 = xr[q];
                acc += a4.x * x4.x + a4.y * x4.y + a4.z * x4.z + a4.w * x4.w;
            }
            out[(size_t)b * T_ + t] = acc;                 // context_vector
        }
    }
}

extern "C" void kernel_launch(void* const* d_in, const int* in_sizes, int n_in,
                              void* d_out, int out_size, void* d_ws, size_t ws_size,
                              hipStream_t stream) {
    const float* x   = (const float*)d_in[0];
    const float* hid = (const float*)d_in[1];
    const float* cel = (const float*)d_in[2];
    const float* w1  = (const float*)d_in[3];
    const float* w1b = (const float*)d_in[4];
    const float* w2  = (const float*)d_in[5];
    const float* w2b = (const float*)d_in[6];
    const float* vk  = (const float*)d_in[7];
    const float* vb  = (const float*)d_in[8];
    float* out = (float*)d_out;

    (void)in_sizes; (void)n_in; (void)out_size; (void)d_ws; (void)ws_size;

    InputAttention_48301202210901_kernel<<<B_, 256, 0, stream>>>(
        x, hid, cel, w1, w1b, w2, w2b, vk, vb, out);
}
